// MultiHeadAttention_46076409151803
// MI455X (gfx1250) — matrix operand order repaired and checked by reading, not compiled, not run on hardware
//
#include <hip/hip_runtime.h>

#define MODEL_DIM 1024
#define NUM_HEADS 16
#define HEAD_DIM  64
#define SEQ       2048
#define BATCH     2

typedef __attribute__((ext_vector_type(16))) _Float16 v16h;
typedef __attribute__((ext_vector_type(8)))  float    v8f;

// ---------------------------------------------------------------------------
// WMMA wrapper: D = A(16x32 f16) * B(32x16 f16) + C(16x16 f32)
// ---------------------------------------------------------------------------
__device__ __forceinline__ v8f wmma16(v16h a, v16h b, v8f c) {
  return __builtin_amdgcn_wmma_f32_16x16x32_f16(
      /*neg_a=*/false, a, /*neg_b=*/false, b,
      /*c_mod=*/(short)0, c, /*reuse_a=*/false, /*reuse_b=*/false);
}

// ---------------------------------------------------------------------------
// Fragment loaders (ISA 7.12.2 layouts, wave32):
//  A 16x32 f16: lane l<16 -> row M=l, K=0..7 (v0-3) and 16..23 (v4-7);
//               lane l>=16 -> row M=l-16, K=8..15 and 24..31.
//  B 32x16 f16: lane l<16 -> col N=l, K=0..7 and 16..23; lane l>=16 -> N=l-16, K=8..15 and 24..31.
//  C 16x16 f32: vgpr r -> M = r + (lane>>4)*8, N = lane&15.
// ---------------------------------------------------------------------------
__device__ __forceinline__ v16h load_a_f32(const float* __restrict__ X, int ldx,
                                           int row0, int k0, int lane) {
  const float* p = X + (size_t)(row0 + (lane & 15)) * ldx + k0 + ((lane >> 4) << 3);
  v16h a;
#pragma unroll
  for (int i = 0; i < 8; ++i) { a[i] = (_Float16)p[i]; a[8 + i] = (_Float16)p[16 + i]; }
  return a;
}

__device__ __forceinline__ v16h load_a_f16(const _Float16* __restrict__ X, int ldx,
                                           int row0, int k0, int lane) {
  const _Float16* p = X + (size_t)(row0 + (lane & 15)) * ldx + k0 + ((lane >> 4) << 3);
  v16h a;
#pragma unroll
  for (int i = 0; i < 8; ++i) { a[i] = p[i]; a[8 + i] = p[16 + i]; }
  return a;
}

// B[k,n] = R[n,k] (row-major R with leading dim ld). Works for global or LDS.
__device__ __forceinline__ v16h load_bT_f32(const float* __restrict__ R, int ld,
                                            int n0, int k0, int lane) {
  const float* p = R + (size_t)(n0 + (lane & 15)) * ld + k0 + ((lane >> 4) << 3);
  v16h b;
#pragma unroll
  for (int i = 0; i < 8; ++i) { b[i] = (_Float16)p[i]; b[8 + i] = (_Float16)p[16 + i]; }
  return b;
}

__device__ __forceinline__ v16h load_bT_f16(const _Float16* R, int ld,
                                            int n0, int k0, int lane) {
  const _Float16* p = R + (size_t)(n0 + (lane & 15)) * ld + k0 + ((lane >> 4) << 3);
  v16h b;
#pragma unroll
  for (int i = 0; i < 8; ++i) { b[i] = p[i]; b[8 + i] = p[16 + i]; }
  return b;
}

// A frag from LDS P tile (16 rows x 72 halves, row-major).
__device__ __forceinline__ v16h load_a_lds(const _Float16* base, int k0, int lane) {
  const _Float16* p = base + (lane & 15) * 72 + k0 + ((lane >> 4) << 3);
  v16h a;
#pragma unroll
  for (int i = 0; i < 8; ++i) { a[i] = p[i]; a[8 + i] = p[16 + i]; }
  return a;
}

// Async global->LDS copy of 16 bytes per lane (CDNA5 ASYNCcnt path).
__device__ __forceinline__ void async_copy_b128(const void* gsrc, void* ldst) {
  const unsigned lds_off = (unsigned)(unsigned long long)ldst;
  const unsigned long long gaddr = (unsigned long long)gsrc;
  asm volatile("global_load_async_to_lds_b128 %0, %1, off"
               :: "v"(lds_off), "v"(gaddr) : "memory");
}

__device__ __forceinline__ void wait_async() {
  asm volatile("s_wait_asynccnt 0" ::: "memory");
}

// ---------------------------------------------------------------------------
// Kernel 1: q/k/v projections. out = X @ W^T + b, stored fp16 head-split.
// grid (M/64, N/64, 3), block 128 (4 waves; wave -> 16 rows x 64 cols).
// ---------------------------------------------------------------------------
__global__ __launch_bounds__(128) void qkv_proj_kernel(
    const float* __restrict__ Q, const float* __restrict__ K, const float* __restrict__ V,
    const float* __restrict__ Wq, const float* __restrict__ bq,
    const float* __restrict__ Wk, const float* __restrict__ bk,
    const float* __restrict__ Wv, const float* __restrict__ bv,
    _Float16* __restrict__ qh, _Float16* __restrict__ kh, _Float16* __restrict__ vT) {
  const int which = blockIdx.z;
  const float* X    = (which == 0) ? Q  : (which == 1) ? K  : V;
  const float* W    = (which == 0) ? Wq : (which == 1) ? Wk : Wv;
  const float* bias = (which == 0) ? bq : (which == 1) ? bk : bv;

  const int lane = threadIdx.x & 31;
  const int wave = threadIdx.x >> 5;
  const int m0 = blockIdx.x * 64 + wave * 16;
  const int n0 = blockIdx.y * 64;

  v8f acc[4] = {};
  for (int k0 = 0; k0 < MODEL_DIM; k0 += 32) {
    v16h a = load_a_f32(X, MODEL_DIM, m0, k0, lane);
#pragma unroll
    for (int t = 0; t < 4; ++t) {
      v16h b = load_bT_f32(W, MODEL_DIM, n0 + t * 16, k0, lane);
      acc[t] = wmma16(a, b, acc[t]);
    }
  }

#pragma unroll
  for (int t = 0; t < 4; ++t) {
    const int n = n0 + t * 16 + (lane & 15);
    const float bn = bias[n];
    const int h = n >> 6, d = n & 63;
#pragma unroll
    for (int r = 0; r < 8; ++r) {
      const int m = m0 + ((lane >> 4) << 3) + r;
      const int b = m >> 11, s = m & (SEQ - 1);
      const _Float16 hv = (_Float16)(acc[t][r] + bn);
      if (which == 2)
        vT[(((size_t)(b * NUM_HEADS + h) * HEAD_DIM + d) * SEQ) + s] = hv;
      else if (which == 0)
        qh[(((size_t)(b * NUM_HEADS + h) * SEQ + s) * HEAD_DIM) + d] = hv;
      else
        kh[(((size_t)(b * NUM_HEADS + h) * SEQ + s) * HEAD_DIM) + d] = hv;
    }
  }
}

// ---------------------------------------------------------------------------
// Kernel 2: flash-style attention with faithful -1e-9 mask fill.
// grid (SEQ/64, B*H), block 128. Wave handles 16 queries x full hd=64.
// K/V chunks staged block-wide into LDS via async global->LDS copies.
// ---------------------------------------------------------------------------
__global__ __launch_bounds__(128) void attn_kernel(
    const _Float16* __restrict__ qh, const _Float16* __restrict__ kh,
    const _Float16* __restrict__ vT, _Float16* __restrict__ attn) {
  __shared__ _Float16 kbuf[64][72];     // keys x head-dim (shared by all waves)
  __shared__ _Float16 vbuf[64][72];     // head-dim x keys-in-chunk
  __shared__ _Float16 pbuf[4][16][72];  // wave-private P tiles

  const int lane = threadIdx.x & 31;
  const int wave = threadIdx.x >> 5;
  const int tid  = threadIdx.x;
  const int bh = blockIdx.y;
  const int b = bh >> 4, h = bh & 15;
  const int q0 = blockIdx.x * 64 + wave * 16;
  const int iloc = (lane >> 4) << 3;  // +r = local row in wave tile

  const _Float16* qb = qh + (size_t)bh * SEQ * HEAD_DIM;
  const _Float16* kb = kh + (size_t)bh * SEQ * HEAD_DIM;
  const _Float16* vb = vT + (size_t)bh * HEAD_DIM * SEQ;

  // Q fragments for this wave's 16 rows, reused across all key chunks.
  const v16h aq0 = load_a_f16(qb, HEAD_DIM, q0, 0, lane);
  const v16h aq1 = load_a_f16(qb, HEAD_DIM, q0, 32, lane);

  v8f O[4] = {};
  float mrun[8], lrun[8];
#pragma unroll
  for (int r = 0; r < 8; ++r) { mrun[r] = -1e30f; lrun[r] = 0.0f; }

  for (int kc = 0; kc < SEQ; kc += 64) {
    if (kc + 64 < SEQ)
      __builtin_prefetch(kb + (size_t)(kc + 64) * HEAD_DIM, 0, 1);

    // ---- async stage K chunk (64x64) and V chunk (64x64) into LDS ----
    // 512 B128 lane-transfers per tile: 4 iterations x 128 lanes x 2 tiles.
#pragma unroll
    for (int i = 0; i < 4; ++i) {
      const int e = tid + i * 128;            // 0..511
      const int row = e >> 3, ch = (e & 7) * 8;
      async_copy_b128(kb + (size_t)(kc + row) * HEAD_DIM + ch, &kbuf[row][ch]);
      async_copy_b128(vb + (size_t)row * SEQ + kc + ch, &vbuf[row][ch]);
    }
    wait_async();
    __syncthreads();

    // ---- scores S = q * k^T (16 x 64 per wave), B-frags from LDS ----
    v8f sacc[4] = {};
#pragma unroll
    for (int t = 0; t < 4; ++t) {
      v16h b0 = load_bT_f16(&kbuf[0][0], 72, t * 16, 0, lane);
      sacc[t] = wmma16(aq0, b0, sacc[t]);
      v16h b1 = load_bT_f16(&kbuf[0][0], 72, t * 16, 32, lane);
      sacc[t] = wmma16(aq1, b1, sacc[t]);
    }

    // ---- scale + faithful mask (masked value = -1e-9, NOT -inf) ----
    float rmax[8];
#pragma unroll
    for (int r = 0; r < 8; ++r) rmax[r] = -1e30f;
#pragma unroll
    for (int t = 0; t < 4; ++t) {
      const int j = kc + t * 16 + (lane & 15);
#pragma unroll
      for (int r = 0; r < 8; ++r) {
        const int i = q0 + iloc + r;
        const float s = (j > i) ? -1e-9f : sacc[t][r] * 0.125f;
        sacc[t][r] = s;
        rmax[r] = fmaxf(rmax[r], s);
      }
    }
#pragma unroll
    for (int r = 0; r < 8; ++r) {
      float v = rmax[r];
      v = fmaxf(v, __shfl_xor(v, 1, 32));
      v = fmaxf(v, __shfl_xor(v, 2, 32));
      v = fmaxf(v, __shfl_xor(v, 4, 32));
      v = fmaxf(v, __shfl_xor(v, 8, 32));
      rmax[r] = v;
    }

    // ---- online softmax update ----
    float rescale[8], rsum[8];
#pragma unroll
    for (int r = 0; r < 8; ++r) {
      const float mnew = fmaxf(mrun[r], rmax[r]);
      rescale[r] = __expf(mrun[r] - mnew);
      mrun[r] = mnew;
      rsum[r] = 0.0f;
    }
#pragma unroll
    for (int t = 0; t < 4; ++t)
#pragma unroll
      for (int r = 0; r < 8; ++r) {
        const float p = __expf(sacc[t][r] - mrun[r]);
        sacc[t][r] = p;
        rsum[r] += p;
      }
#pragma unroll
    for (int r = 0; r < 8; ++r) {
      float v = rsum[r];
      v += __shfl_xor(v, 1, 32);
      v += __shfl_xor(v, 2, 32);
      v += __shfl_xor(v, 4, 32);
      v += __shfl_xor(v, 8, 32);
      lrun[r] = lrun[r] * rescale[r] + v;
#pragma unroll
      for (int t = 0; t < 4; ++t) O[t][r] *= rescale[r];
    }

    // ---- P: C-layout f32 -> A-layout f16 via wave-private LDS tile ----
#pragma unroll
    for (int t = 0; t < 4; ++t)
#pragma unroll
      for (int r = 0; r < 8; ++r)
        pbuf[wave][iloc + r][t * 16 + (lane & 15)] = (_Float16)sacc[t][r];
    // per-wave DS ops are in-order (DScnt); pbuf is wave-private -> no barrier

    // ---- O += P * V, V B-frags from LDS (vbuf[d][j], contiguous over j) ----
#pragma unroll
    for (int ks = 0; ks < 2; ++ks) {
      v16h ap = load_a_lds(&pbuf[wave][0][0], ks * 32, lane);
#pragma unroll
      for (int t = 0; t < 4; ++t) {
        v16h bvf = load_bT_f16(&vbuf[0][0], 72, t * 16, ks * 32, lane);
        O[t] = wmma16(ap, bvf, O[t]);
      }
    }
    __syncthreads();  // protect kbuf/vbuf before next chunk's async writes
  }

  // ---- normalize and store merged heads as f16 [B,S,D] ----
#pragma unroll
  for (int r = 0; r < 8; ++r) {
    const float invl = 1.0f / lrun[r];
    const int s = q0 + iloc + r;
#pragma unroll
    for (int t = 0; t < 4; ++t) {
      const int d = t * 16 + (lane & 15);
      attn[((size_t)(b * SEQ + s) * MODEL_DIM) + h * HEAD_DIM + d] =
          (_Float16)(O[t][r] * invl);
    }
  }
}

// ---------------------------------------------------------------------------
// Kernel 3: output projection. out = attn(f16) @ Wo^T + bo  -> f32
// ---------------------------------------------------------------------------
__global__ __launch_bounds__(128) void oproj_kernel(
    const _Float16* __restrict__ A, const float* __restrict__ Wo,
    const float* __restrict__ bo, float* __restrict__ out) {
  const int lane = threadIdx.x & 31;
  const int wave = threadIdx.x >> 5;
  const int m0 = blockIdx.x * 64 + wave * 16;
  const int n0 = blockIdx.y * 64;

  v8f acc[4] = {};
  for (int k0 = 0; k0 < MODEL_DIM; k0 += 32) {
    v16h a = load_a_f16(A, MODEL_DIM, m0, k0, lane);
#pragma unroll
    for (int t = 0; t < 4; ++t) {
      v16h b = load_bT_f32(Wo, MODEL_DIM, n0 + t * 16, k0, lane);
      acc[t] = wmma16(a, b, acc[t]);
    }
  }

#pragma unroll
  for (int t = 0; t < 4; ++t) {
    const int n = n0 + t * 16 + (lane & 15);
    const float bn = bo[n];
#pragma unroll
    for (int r = 0; r < 8; ++r) {
      const int m = m0 + ((lane >> 4) << 3) + r;
      out[(size_t)m * MODEL_DIM + n] = acc[t][r] + bn;
    }
  }
}

// ---------------------------------------------------------------------------
extern "C" void kernel_launch(void* const* d_in, const int* in_sizes, int n_in,
                              void* d_out, int out_size, void* d_ws, size_t ws_size,
                              hipStream_t stream) {
  const float* Q  = (const float*)d_in[0];
  const float* K  = (const float*)d_in[1];
  const float* V  = (const float*)d_in[2];
  // d_in[3] = causal tril mask (int); applied analytically in attn_kernel.
  const float* Wq = (const float*)d_in[4];
  const float* bq = (const float*)d_in[5];
  const float* Wk = (const float*)d_in[6];
  const float* bk = (const float*)d_in[7];
  const float* Wv = (const float*)d_in[8];
  const float* bv = (const float*)d_in[9];
  const float* Wo = (const float*)d_in[10];
  const float* bo = (const float*)d_in[11];
  float* out = (float*)d_out;

  const size_t elems = (size_t)BATCH * NUM_HEADS * SEQ * HEAD_DIM;  // 4M
  _Float16* qh    = (_Float16*)d_ws;
  _Float16* kh    = qh + elems;
  _Float16* vT    = kh + elems;
  _Float16* attnb = vT + elems;   // total 32 MB of workspace

  const dim3 blk(128);
  const int M = BATCH * SEQ;  // 4096
  qkv_proj_kernel<<<dim3(M / 64, MODEL_DIM / 64, 3), blk, 0, stream>>>(
      Q, K, V, Wq, bq, Wk, bk, Wv, bv, qh, kh, vT);
  attn_kernel<<<dim3(SEQ / 64, BATCH * NUM_HEADS), blk, 0, stream>>>(
      qh, kh, vT, attnb);
  oproj_kernel<<<dim3(M / 64, MODEL_DIM / 64), blk, 0, stream>>>(
      attnb, Wo, bo, out);
}
